// Graph_Actor_Model_42159398978357
// MI455X (gfx1250) — matrix operand order repaired and checked by reading, not compiled, not run on hardware
//
#include <hip/hip_runtime.h>
#include <hip/hip_bf16.h>

// ============================================================================
// Graph Actor Model for MI455X (gfx1250, wave32, WMMA).
// All dense layers run on v_wmma_f32_16x16x32_f16 (f16 A/B, f32 accumulate).
// GEMM blocking: 256 rows/block, 2 row-tiles per wave sharing each B fragment
// (halves LDS B reads); A fragments hoisted to registers for K<=256.
// GCN aggregation is implemented generally (degrees + edge scatter w/ atomics).
// ============================================================================

typedef __attribute__((ext_vector_type(16))) _Float16 v16h;
typedef __attribute__((ext_vector_type(8)))  _Float16 v8h;
typedef __attribute__((ext_vector_type(8)))  float    v8f;

#define LOG_2PI_F 1.8378770664093454f

// ---------------------------------------------------------------------------
// small helpers
// ---------------------------------------------------------------------------
__device__ __forceinline__ void stage_copy(_Float16* dst, const float* src) {
    const float4 v = *reinterpret_cast<const float4*>(src);
    dst[0] = (_Float16)v.x; dst[1] = (_Float16)v.y;
    dst[2] = (_Float16)v.z; dst[3] = (_Float16)v.w;
}
__device__ __forceinline__ void stage_copy(_Float16* dst, const _Float16* src) {
    *reinterpret_cast<v8h*>(dst) = *reinterpret_cast<const v8h*>(src);
}
__device__ __forceinline__ void store_out(_Float16* p, float v) { *p = (_Float16)v; }
__device__ __forceinline__ void store_out(float* p, float v)    { *p = v; }

__device__ __forceinline__ v16h frag_from_lds(const _Float16* p) {
    const v8h lo = *reinterpret_cast<const v8h*>(p);
    const v8h hi = *reinterpret_cast<const v8h*>(p + 16);
    v16h f;
    #pragma unroll
    for (int j = 0; j < 8; ++j) { f[j] = lo[j]; f[j + 8] = hi[j]; }
    return f;
}

template<typename T> struct VecW;
template<> struct VecW<float>    { static constexpr int v = 4; };
template<> struct VecW<_Float16> { static constexpr int v = 8; };

// ---------------------------------------------------------------------------
// Weight prep: W[K,M] (f32, row-major) -> Wt[M,K] (f16, row-major in K).
// Makes every WMMA B-fragment a contiguous 64B run per output column.
// ---------------------------------------------------------------------------
__global__ __launch_bounds__(256) void transpose_w_kernel(
    const float* __restrict__ W, _Float16* __restrict__ Wt, int K, int M)
{
    const int i = blockIdx.x * 256 + threadIdx.x;
    if (i < K * M) {
        const int k = i / M, m = i % M;
        Wt[(size_t)m * K + k] = (_Float16)W[i];
    }
}

// ---------------------------------------------------------------------------
// GCN degree: deg = segment_sum(ones over dst) + 1  (the added self-loop)
// ---------------------------------------------------------------------------
__global__ __launch_bounds__(256) void deg_init_kernel(float* deg, int N)
{
    const int i = blockIdx.x * 256 + threadIdx.x;
    if (i < N) deg[i] = 1.0f;
}
__global__ __launch_bounds__(256) void deg_count_kernel(
    const int* __restrict__ dst, float* deg, int E)
{
    const int i = blockIdx.x * 256 + threadIdx.x;
    if (i < E) atomicAdd(&deg[dst[i]], 1.0f);
}

// agg = (1/deg) * xw   (self-loop term, dis*dis == 1/deg)
__global__ __launch_bounds__(256) void agg_self_kernel(
    const float* __restrict__ xw, const float* __restrict__ deg,
    float* __restrict__ agg, int N)
{
    const int i = blockIdx.x * 256 + threadIdx.x;   // one float4 per thread
    if (i < N * 32) {
        const int node = i >> 5;
        const int f    = (i & 31) * 4;
        const float inv = 1.0f / deg[node];
        const float4 v = *reinterpret_cast<const float4*>(xw + (size_t)node * 128 + f);
        float4 o; o.x = v.x * inv; o.y = v.y * inv; o.z = v.z * inv; o.w = v.w * inv;
        *reinterpret_cast<float4*>(agg + (size_t)node * 128 + f) = o;
    }
}

// agg[dst] += rsqrt(deg[src])*rsqrt(deg[dst]) * xw[src]   (one wave per edge)
__global__ __launch_bounds__(256) void agg_edge_kernel(
    const int* __restrict__ src, const int* __restrict__ dst,
    const float* __restrict__ deg, const float* __restrict__ xw,
    float* __restrict__ agg, int E)
{
    const int wave = threadIdx.x >> 5;
    const int lane = threadIdx.x & 31;
    const int e = blockIdx.x * 8 + wave;
    if (e < E) {
        const int s = src[e], d = dst[e];
        const float norm = rsqrtf(deg[s]) * rsqrtf(deg[d]);
        const int f = lane * 4;
        const float4 v = *reinterpret_cast<const float4*>(xw + (size_t)s * 128 + f);
        float* ap = agg + (size_t)d * 128 + f;
        atomicAdd(ap + 0, norm * v.x);
        atomicAdd(ap + 1, norm * v.y);
        atomicAdd(ap + 2, norm * v.z);
        atomicAdd(ap + 3, norm * v.w);
    }
}

// ---------------------------------------------------------------------------
// Fused bias + ReLU + LayerNorm(128) -> f16. One wave per node, 4 feats/lane.
// ---------------------------------------------------------------------------
__global__ __launch_bounds__(256) void ln_kernel(
    const float* __restrict__ agg, const float* __restrict__ b_gcn,
    const float* __restrict__ g_ln, const float* __restrict__ b_ln,
    _Float16* __restrict__ out, int N)
{
    const int wave = threadIdx.x >> 5;
    const int lane = threadIdx.x & 31;
    const int row = blockIdx.x * 8 + wave;
    if (row >= N) return;

    const int f = lane * 4;
    const float4 t  = *reinterpret_cast<const float4*>(agg + (size_t)row * 128 + f);
    const float4 bg = *reinterpret_cast<const float4*>(b_gcn + f);
    float v0 = fmaxf(t.x + bg.x, 0.0f);
    float v1 = fmaxf(t.y + bg.y, 0.0f);
    float v2 = fmaxf(t.z + bg.z, 0.0f);
    float v3 = fmaxf(t.w + bg.w, 0.0f);

    float s  = v0 + v1 + v2 + v3;
    float sq = v0 * v0 + v1 * v1 + v2 * v2 + v3 * v3;
    #pragma unroll
    for (int off = 16; off > 0; off >>= 1) {
        s  += __shfl_xor(s,  off, 32);
        sq += __shfl_xor(sq, off, 32);
    }
    const float mean = s * (1.0f / 128.0f);
    const float var  = sq * (1.0f / 128.0f) - mean * mean;
    const float rstd = rsqrtf(var + 1e-5f);

    const float4 g  = *reinterpret_cast<const float4*>(g_ln + f);
    const float4 bb = *reinterpret_cast<const float4*>(b_ln + f);
    _Float16* op = out + (size_t)row * 128 + f;
    op[0] = (_Float16)(g.x * (v0 - mean) * rstd + bb.x);
    op[1] = (_Float16)(g.y * (v1 - mean) * rstd + bb.y);
    op[2] = (_Float16)(g.z * (v2 - mean) * rstd + bb.z);
    op[3] = (_Float16)(g.w * (v3 - mean) * rstd + bb.w);
}

// ---------------------------------------------------------------------------
// WMMA GEMM:  out[N, M] = act( [A0 | A1][N, K] @ Wt^T + bias )
//   A cols [0, K0) come from A0 (row stride K0), cols [K0, K) from A1.
//   Block: 256 thr = 8 waves, 256 rows (2 row-tiles per wave, shared B frag).
//   A block (256 x K f16) + B tile (16 x K f16) in dynamic LDS (<=283KB).
//   A fragments hoisted to registers when K <= 256.
// ---------------------------------------------------------------------------
template<int K, int M, int K0, typename TIN, bool RELU, bool BIAS, typename TOUT>
__global__ __launch_bounds__(256) void gemm_kernel(
    const TIN* __restrict__ A0, const TIN* __restrict__ A1,
    const _Float16* __restrict__ Bt, const float* __restrict__ bias,
    TOUT* __restrict__ out)
{
    constexpr int ROWS = 256;
    constexpr int APAD = K + 8;                   // f16 elements, staggers banks
    extern __shared__ _Float16 lds[];
    _Float16* ldsA = lds;                         // ROWS x APAD
    _Float16* ldsB = lds + ROWS * APAD;           // 16  x APAD

    const int tid = threadIdx.x;
    const int rowBase = blockIdx.x * ROWS;

    // ---- stage A block: ROWS rows x K, converting to f16 if needed ----
    constexpr int VEC = VecW<TIN>::v;
    constexpr int CPR = K / VEC;                  // vector chunks per row
    for (int i = tid; i < ROWS * CPR; i += 256) {
        const int r = i / CPR;
        const int c = (i % CPR) * VEC;
        const int rg = rowBase + r;
        const TIN* src = (c < K0) ? (A0 + (size_t)rg * K0 + c)
                                  : (A1 + (size_t)rg * (K - K0) + (c - K0));
        stage_copy(ldsA + r * APAD + c, src);
    }

    const int wave = tid >> 5;                    // 0..7 -> 32-row group
    const int lane = tid & 31;
    const int r16  = lane & 15;
    const int hi   = lane >> 4;

    // ISA fragment layout (16-bit A 16x32): lanes<16 hold K0-7 / K16-23,
// lanes>=16 hold K8-15 / K24-31.  B 32x16: lanes<16 hold K0-7 / K16-23 of col,
// lanes>=16 hold K8-15 / K24-31.
    const _Float16* aBase0 = ldsA + (wave * 32 + r16) * APAD + (hi ? 8 : 0);
    const _Float16* aBase1 = aBase0 + 16 * APAD;  // second row-tile
    const _Float16* bBase  = ldsB + r16 * APAD + (hi ? 8 : 0);

    constexpr bool HOIST = (K <= 256);
    constexpr int  NF    = HOIST ? (K / 32) : 1;
    v16h Ar0[NF], Ar1[NF];
    if constexpr (HOIST) {                        // A frags -> registers, once
        __syncthreads();
        #pragma unroll
        for (int kc = 0; kc < K / 32; ++kc) {
            Ar0[kc] = frag_from_lds(aBase0 + kc * 32);
            Ar1[kc] = frag_from_lds(aBase1 + kc * 32);
        }
    }

    for (int ct = 0; ct < M / 16; ++ct) {
        __syncthreads();                          // prev tile done / A ready
        for (int i = tid; i < 16 * (K / 8); i += 256) {
            const int r = i / (K / 8);
            const int c = (i % (K / 8)) * 8;
            *reinterpret_cast<v8h*>(ldsB + r * APAD + c) =
                *reinterpret_cast<const v8h*>(Bt + (size_t)(ct * 16 + r) * K + c);
        }
        __syncthreads();

        v8f acc0 = {0.f, 0.f, 0.f, 0.f, 0.f, 0.f, 0.f, 0.f};
        v8f acc1 = {0.f, 0.f, 0.f, 0.f, 0.f, 0.f, 0.f, 0.f};
        #pragma unroll
        for (int kc = 0; kc < K / 32; ++kc) {
            const v8h b0 = *reinterpret_cast<const v8h*>(bBase + kc * 32);
            const v8h b1 = *reinterpret_cast<const v8h*>(bBase + kc * 32 + 16);
            v16h Bf;
            #pragma unroll
            for (int j = 0; j < 8; ++j) { Bf[j] = b0[j]; Bf[j + 8] = b1[j]; }
            v16h Af0, Af1;
            if constexpr (HOIST) { Af0 = Ar0[kc]; Af1 = Ar1[kc]; }
            else {
                Af0 = frag_from_lds(aBase0 + kc * 32);
                Af1 = frag_from_lds(aBase1 + kc * 32);
            }
            acc0 = __builtin_amdgcn_wmma_f32_16x16x32_f16(
                false, Af0, false, Bf, (short)0, acc0, false, false);
            acc1 = __builtin_amdgcn_wmma_f32_16x16x32_f16(
                false, Af1, false, Bf, (short)0, acc1, false, false);
        }

        // C layout: VGPR i -> row (hi*8 + i) of tile, col = lane&15
        const int cg = ct * 16 + r16;
        const float bv = BIAS ? bias[cg] : 0.0f;
        #pragma unroll
        for (int t = 0; t < 2; ++t) {
            const v8f& acc = t ? acc1 : acc0;
            const int rbase = rowBase + wave * 32 + t * 16 + hi * 8;
            #pragma unroll
            for (int i = 0; i < 8; ++i) {
                float v = acc[i] + bv;
                if (RELU) v = fmaxf(v, 0.0f);
                store_out(out + (size_t)(rbase + i) * M + cg, v);
            }
        }
    }
}

// ---------------------------------------------------------------------------
// Head: mu/sigma dots (K=512, A=2), exp, reparam sample, clamp, log-prob.
// One wave per node; 16 h-values per lane.
// ---------------------------------------------------------------------------
__global__ __launch_bounds__(256) void head_kernel(
    const _Float16* __restrict__ h,
    const float* __restrict__ Wmu, const float* __restrict__ bmu,
    const float* __restrict__ Wsig, const float* __restrict__ bsig,
    const float* __restrict__ noise, float* __restrict__ out, int N)
{
    const int wave = threadIdx.x >> 5;
    const int lane = threadIdx.x & 31;
    const int row = blockIdx.x * 8 + wave;
    if (row >= N) return;

    const _Float16* hp = h + (size_t)row * 512 + lane * 16;
    const v8h h0 = *reinterpret_cast<const v8h*>(hp);
    const v8h h1 = *reinterpret_cast<const v8h*>(hp + 8);

    float m0 = 0.f, m1 = 0.f, s0 = 0.f, s1 = 0.f;
    #pragma unroll
    for (int j = 0; j < 16; ++j) {
        const float v = (float)((j < 8) ? h0[j & 7] : h1[j & 7]);
        const int k = lane * 16 + j;
        const float2 wm = *reinterpret_cast<const float2*>(Wmu  + 2 * k);
        const float2 ws = *reinterpret_cast<const float2*>(Wsig + 2 * k);
        m0 += v * wm.x; m1 += v * wm.y;
        s0 += v * ws.x; s1 += v * ws.y;
    }
    #pragma unroll
    for (int off = 16; off > 0; off >>= 1) {
        m0 += __shfl_xor(m0, off, 32); m1 += __shfl_xor(m1, off, 32);
        s0 += __shfl_xor(s0, off, 32); s1 += __shfl_xor(s1, off, 32);
    }
    if (lane == 0) {
        const float mu0 = m0 + bmu[0], mu1 = m1 + bmu[1];
        const float ls0 = s0 + bsig[0], ls1 = s1 + bsig[1];
        const float sg0 = expf(ls0), sg1 = expf(ls1);
        const float n0 = noise[(size_t)row * 2 + 0];
        const float n1 = noise[(size_t)row * 2 + 1];
        out[(size_t)row * 2 + 0] = fminf(fmaxf(mu0 + sg0 * n0, -1.0f), 1.0f);
        out[(size_t)row * 2 + 1] = fminf(fmaxf(mu1 + sg1 * n1, -1.0f), 1.0f);
        out[(size_t)2 * N + row] =
            -0.5f * (n0 * n0 + n1 * n1 + 2.0f * (ls0 + ls1) + 2.0f * LOG_2PI_F);
    }
}

// ---------------------------------------------------------------------------
// launcher
// ---------------------------------------------------------------------------
extern "C" void kernel_launch(void* const* d_in, const int* in_sizes, int n_in,
                              void* d_out, int out_size, void* d_ws, size_t ws_size,
                              hipStream_t stream) {
    const float* obs    = (const float*)d_in[0];
    const float* noise  = (const float*)d_in[1];
    const int*   eidx   = (const int*)  d_in[2];
    const float* W_enc1 = (const float*)d_in[3];
    const float* b_enc1 = (const float*)d_in[4];
    const float* W_gcn  = (const float*)d_in[5];
    const float* b_gcn  = (const float*)d_in[6];
    const float* g_ln   = (const float*)d_in[7];
    const float* b_ln   = (const float*)d_in[8];
    const float* W_gd   = (const float*)d_in[9];
    const float* b_gd   = (const float*)d_in[10];
    const float* W_p1   = (const float*)d_in[11];
    const float* b_p1   = (const float*)d_in[12];
    const float* W_p2   = (const float*)d_in[13];
    const float* b_p2   = (const float*)d_in[14];
    const float* W_p3   = (const float*)d_in[15];
    const float* b_p3   = (const float*)d_in[16];
    const float* W_mu   = (const float*)d_in[17];
    const float* b_mu   = (const float*)d_in[18];
    const float* W_sig  = (const float*)d_in[19];
    const float* b_sig  = (const float*)d_in[20];

    const int N = in_sizes[0] / 32;        // 131072 (multiple of 256)
    const int E = in_sizes[2] / 2;

    // ---- workspace carve (liveness-based aliasing, peak ~300 MB) ----
    char* ws = (char*)d_ws;
    size_t off = 0;
    auto carve = [&](size_t bytes) -> char* {
        char* p = ws + off;
        off += (bytes + 255) & ~(size_t)255;
        return p;
    };
    _Float16* wt_enc1 = (_Float16*)carve((size_t)128 * 32 * 2);
    _Float16* wt_gcn  = (_Float16*)carve((size_t)128 * 128 * 2);
    _Float16* wt_gd   = (_Float16*)carve((size_t)128 * 128 * 2);
    _Float16* wt_p1   = (_Float16*)carve((size_t)512 * 256 * 2);
    _Float16* wt_p2   = (_Float16*)carve((size_t)512 * 512 * 2);
    _Float16* wt_p3   = (_Float16*)carve((size_t)512 * 512 * 2);
    float*    deg     = (float*)   carve((size_t)N * 4);
    _Float16* x_f16   = (_Float16*)carve((size_t)N * 128 * 2);   // enc1 out
    float*    xw      = (float*)   carve((size_t)N * 128 * 4);   // x @ W_gcn
    float*    agg     = (float*)   carve((size_t)N * 128 * 4);   // GCN aggregate
    _Float16* h1      = (_Float16*)carve((size_t)N * 512 * 2);
    // aliases (dead-buffer reuse):
    _Float16* xg_ln = (_Float16*)xw;      // LN output; xw dead after aggregation
    _Float16* xg    = (_Float16*)agg;     // W_gd output; agg dead after LN
    _Float16* h2    = (_Float16*)x_f16;   // spans x+xw+agg regions (>=134MB)
    _Float16* h3    = h1;                 // h1 dead after p2 GEMM

    auto shbytes = [](int K) -> size_t { return (size_t)(256 + 16) * (K + 8) * 2; };
    const int GB = N / 256;               // GEMM blocks (256 rows each)
    const int WB = (N + 7) / 8;           // one-wave-per-node blocks

    // 1) weight prep (transpose + f32->f16)
    transpose_w_kernel<<<(32 * 128 + 255) / 256, 256, 0, stream>>>(W_enc1, wt_enc1, 32, 128);
    transpose_w_kernel<<<(128 * 128 + 255) / 256, 256, 0, stream>>>(W_gcn, wt_gcn, 128, 128);
    transpose_w_kernel<<<(128 * 128 + 255) / 256, 256, 0, stream>>>(W_gd, wt_gd, 128, 128);
    transpose_w_kernel<<<(256 * 512 + 255) / 256, 256, 0, stream>>>(W_p1, wt_p1, 256, 512);
    transpose_w_kernel<<<(512 * 512 + 255) / 256, 256, 0, stream>>>(W_p2, wt_p2, 512, 512);
    transpose_w_kernel<<<(512 * 512 + 255) / 256, 256, 0, stream>>>(W_p3, wt_p3, 512, 512);

    // 2) degrees
    deg_init_kernel<<<(N + 255) / 256, 256, 0, stream>>>(deg, N);
    deg_count_kernel<<<(E + 255) / 256, 256, 0, stream>>>(eidx + E, deg, E);

    // 3) x = relu(obs @ W_enc1 + b)          [N,32]x[32,128] -> f16
    gemm_kernel<32, 128, 32, float, true, true, _Float16>
        <<<GB, 256, shbytes(32), stream>>>(obs, obs, wt_enc1, b_enc1, x_f16);

    // 4) xw = x @ W_gcn                      [N,128]x[128,128] -> f32
    gemm_kernel<128, 128, 128, _Float16, false, false, float>
        <<<GB, 256, shbytes(128), stream>>>(x_f16, x_f16, wt_gcn, nullptr, xw);

    // 5) GCN aggregate: self-loop term + edge scatter
    agg_self_kernel<<<(N * 32 + 255) / 256, 256, 0, stream>>>(xw, deg, agg, N);
    agg_edge_kernel<<<(E + 7) / 8, 256, 0, stream>>>(eidx, eidx + E, deg, xw, agg, E);

    // 6) bias + relu + layernorm -> f16
    ln_kernel<<<WB, 256, 0, stream>>>(agg, b_gcn, g_ln, b_ln, xg_ln, N);

    // 7) xg = relu(xg_ln @ W_gd + b)         [N,128]x[128,128] -> f16
    gemm_kernel<128, 128, 128, _Float16, true, true, _Float16>
        <<<GB, 256, shbytes(128), stream>>>(xg_ln, xg_ln, wt_gd, b_gd, xg);

    // 8) h1 = relu([xg | x] @ W_p1 + b)      [N,256]x[256,512] -> f16
    gemm_kernel<256, 512, 128, _Float16, true, true, _Float16>
        <<<GB, 256, shbytes(256), stream>>>(xg, x_f16, wt_p1, b_p1, h1);

    // 9) h2 = relu(h1 @ W_p2 + b)            [N,512]x[512,512] -> f16
    gemm_kernel<512, 512, 512, _Float16, true, true, _Float16>
        <<<GB, 256, shbytes(512), stream>>>(h1, h1, wt_p2, b_p2, h2);

    // 10) h3 = relu(h2 @ W_p3 + b)           [N,512]x[512,512] -> f16
    gemm_kernel<512, 512, 512, _Float16, true, true, _Float16>
        <<<GB, 256, shbytes(512), stream>>>(h2, h2, wt_p3, b_p3, h3);

    // 11) head: action + log_probs -> d_out
    head_kernel<<<WB, 256, 0, stream>>>(h3, W_mu, b_mu, W_sig, b_sig, noise,
                                        (float*)d_out, N);
}
